// GatedAttentionUnit_80461917323567
// MI455X (gfx1250) — matrix operand order repaired and checked by reading, not compiled, not run on hardware
//
#include <hip/hip_runtime.h>
#include <hip/hip_bf16.h>

// ---------------------------------------------------------------------------
// GAU forward for MI455X (gfx1250): 4 bf16 WMMA GEMMs + fused epilogues.
// This revision: double-buffered LDS staging fed by GLOBAL_LOAD_ASYNC_TO_LDS
// (ASYNCcnt-tracked, memory->LDS direct) so global fetch of tile i+1 overlaps
// the WMMA stream on tile i.
// ---------------------------------------------------------------------------

typedef __bf16        bf16x16 __attribute__((ext_vector_type(16)));
typedef float         f32x8   __attribute__((ext_vector_type(8)));
typedef unsigned int  u32x4   __attribute__((ext_vector_type(4)));

union Frag {
  bf16x16 bf;
  u32x4   q[2];
};

__device__ __forceinline__ unsigned short f2bfu(float f) {
  union { float f; unsigned u; } v; v.f = f;
  unsigned u = v.u;
  return (unsigned short)((u + 0x7FFFu + ((u >> 16) & 1u)) >> 16); // RNE
}
__device__ __forceinline__ float bf2f(unsigned short s) {
  union { unsigned u; float f; } v; v.u = ((unsigned)s) << 16;
  return v.f;
}

// Async memory -> LDS, 16B per lane, no VGPR staging. Tracked by ASYNCcnt.
__device__ __forceinline__ void async_b128(unsigned lds_off, const void* gaddr) {
  asm volatile("global_load_async_to_lds_b128 %0, %1, off"
               :: "v"(lds_off), "v"((unsigned long)gaddr) : "memory");
}
__device__ __forceinline__ void wait_async_le4() {
  asm volatile("s_wait_asynccnt 0x4" ::: "memory");
}
__device__ __forceinline__ void wait_async_0() {
  asm volatile("s_wait_asynccnt 0x0" ::: "memory");
}
__device__ __forceinline__ unsigned lds_off_of(const void* p) {
  return (unsigned)(unsigned long)p;   // flat shared addr[31:0] == LDS offset
}

// ---------------------------------------------------------------------------
// Generic 128x128 tile WMMA GEMM. C = A(MxK, row major, bf16) * B^T where the
// second operand is staged N-major: Bt[n][k].  8 waves per block (2 along M x
// 4 along N); each wave owns a 64x32 sub-tile = 4x2 WMMA 16x16 accumulators.
// EPI: 0 = bias+SiLU+split(u/vT/q/k)   1 = scale -> f32
//      2 = gate(u) -> bf16             3 = bias -> f32
// ---------------------------------------------------------------------------
struct GemmP {
  const unsigned short* A;
  const unsigned short* B;
  long sA, sB;                 // per-batch strides (elements)
  int  K, lda, ldb;
  // epilogue state
  float*           outf;  long sOutF; int ldf;
  unsigned short*  outbf; long sOutBF;
  const unsigned short* gate; long sGate;
  const float* bias;
  const float* qw; const float* qb; const float* kw; const float* kb;
  unsigned short* u_out; unsigned short* vt_out;
  unsigned short* q_out; unsigned short* k_out;
  float scale;
};

template<int EPI>
__global__ __launch_bounds__(256) void gau_gemm(GemmP p) {
  constexpr int PITCH = 40;                         // 80B rows: conflict-spread, 16B aligned
  __shared__ __align__(16) unsigned short As[2][128 * PITCH];
  __shared__ __align__(16) unsigned short Bs[2][128 * PITCH];

  const int tid  = threadIdx.x;
  const int wave = tid >> 5, lane = tid & 31;
  const int wm   = wave >> 2, wn = wave & 3;        // 2x4 wave grid
  const int h    = lane >> 4, nl = lane & 15;
  const int z    = blockIdx.z;
  const int bx   = blockIdx.x, by = blockIdx.y;

  const unsigned short* Ag = p.A + (long)z * p.sA + (long)(by * 128) * p.lda;
  const unsigned short* Bg = p.B + (long)z * p.sB + (long)(bx * 128) * p.ldb;

  const int r0 = tid >> 2;                          // 0..63 (row pairs r0, r0+64)
  const int c0 = (tid & 3) * 8;                     // ushort offset in 32-wide row

  // Per-thread LDS landing slots (byte offsets), one per buffer.
  const unsigned laA0 = lds_off_of(&As[0][r0 * PITCH + c0]);
  const unsigned laB0 = lds_off_of(&Bs[0][r0 * PITCH + c0]);
  const unsigned laA1 = lds_off_of(&As[1][r0 * PITCH + c0]);
  const unsigned laB1 = lds_off_of(&Bs[1][r0 * PITCH + c0]);
  constexpr unsigned HALF = 64u * PITCH * 2u;       // +64 rows, in bytes

  // Issue one 128x32 A tile + 128x32 Bt tile into buffer `buf` (4 asyncs/thread).
  auto issue_tile = [&](int buf, int k0) {
    const unsigned short* a0 = Ag + (long)r0        * p.lda + k0 + c0;
    const unsigned short* a1 = Ag + (long)(r0 + 64) * p.lda + k0 + c0;
    const unsigned short* b0 = Bg + (long)r0        * p.ldb + k0 + c0;
    const unsigned short* b1 = Bg + (long)(r0 + 64) * p.ldb + k0 + c0;
    const unsigned lA = buf ? laA1 : laA0;
    const unsigned lB = buf ? laB1 : laB0;
    async_b128(lA,        a0);
    async_b128(lA + HALF, a1);
    async_b128(lB,        b0);
    async_b128(lB + HALF, b1);
  };

  f32x8 acc[4][2];
  #pragma unroll
  for (int i = 0; i < 4; ++i)
    #pragma unroll
    for (int j = 0; j < 2; ++j)
      #pragma unroll
      for (int e = 0; e < 8; ++e) acc[i][j][e] = 0.f;

  const int nIter = p.K >> 5;
  issue_tile(0, 0);

  for (int it = 0; it < nIter; ++it) {
    const int cur = it & 1;
    if (it + 1 < nIter) {
      issue_tile(cur ^ 1, (it + 1) << 5);           // prefetch next tile (async)
      wait_async_le4();                             // in-order: first 4 (cur) done
    } else {
      wait_async_0();
    }
    __syncthreads();                                // all waves: cur tile visible

    const unsigned short* Ac = As[cur];
    const unsigned short* Bc = Bs[cur];
    Frag af[4], bfr[2];
    #pragma unroll
    for (int mi = 0; mi < 4; ++mi) {                // A 16x32 bf16 layout (ISA 7.12.2)
      const int ar = wm * 64 + mi * 16 + nl;
      af[mi].q[0] = *(const u32x4*)&Ac[ar * PITCH + 8 * h];
      af[mi].q[1] = *(const u32x4*)&Ac[ar * PITCH + 16 + 8 * h];
    }
    #pragma unroll
    for (int ni = 0; ni < 2; ++ni) {                // B 32x16: lane half -> K 8h..8h+7, 16+8h..16+8h+7
      const int br = wn * 32 + ni * 16 + nl;
      bfr[ni].q[0] = *(const u32x4*)&Bc[br * PITCH + 8 * h];
      bfr[ni].q[1] = *(const u32x4*)&Bc[br * PITCH + 16 + 8 * h];
    }
    #pragma unroll
    for (int mi = 0; mi < 4; ++mi)
      #pragma unroll
      for (int ni = 0; ni < 2; ++ni)
        acc[mi][ni] = __builtin_amdgcn_wmma_f32_16x16x32_bf16(
            false, af[mi].bf, false, bfr[ni].bf, (short)0, acc[mi][ni], false, false);
    __syncthreads();                                // cur buffer free for reuse
  }

  // Epilogue: VGPR r of each 16x16 tile holds (m = r + 8h, n = lane&15).
  #pragma unroll
  for (int mi = 0; mi < 4; ++mi) {
    #pragma unroll
    for (int ni = 0; ni < 2; ++ni) {
      #pragma unroll
      for (int r = 0; r < 8; ++r) {
        const int m = by * 128 + wm * 64 + mi * 16 + 8 * h + r;
        const int n = bx * 128 + wn * 32 + ni * 16 + nl;
        float c = acc[mi][ni][r];
        if constexpr (EPI == 0) {                   // bias + SiLU + split
          float hv = c + p.bias[n];
          hv = hv / (1.f + __expf(-hv));
          if (n < 2048) {
            p.u_out[(long)m * 2048 + n] = f2bfu(hv);
          } else if (n < 4096) {
            const int b = m >> 11, s = m & 2047, vg = n - 2048;
            p.vt_out[((long)b * 2048 + vg) * 2048 + s] = f2bfu(hv);
          } else {
            const int j = n - 4096;
            p.q_out[(long)m * 128 + j] = f2bfu(hv * p.qw[j] + p.qb[j]);
            p.k_out[(long)m * 128 + j] = f2bfu(hv * p.kw[j] + p.kb[j]);
          }
        } else if constexpr (EPI == 1) {            // scores * 1/sqrt(QK) -> f32
          p.outf[(long)z * p.sOutF + (long)m * p.ldf + n] = c * p.scale;
        } else if constexpr (EPI == 2) {            // u-gate -> bf16
          const long gi = (long)z * p.sGate + (long)m * 2048 + n;
          p.outbf[(long)z * p.sOutBF + (long)m * 2048 + n] =
              f2bfu(c * bf2f(p.gate[gi]));
        } else {                                    // + b_o -> f32 output
          p.outf[(long)m * p.ldf + n] = c + p.bias[n];
        }
      }
    }
  }
}

// ---------------------------------------------------------------------------
// Row softmax over 2048 scores with padding mask; emits bf16 A.
// ---------------------------------------------------------------------------
__global__ __launch_bounds__(256) void gau_softmax(const float* __restrict__ Sc,
                                                   const int* __restrict__ mask,
                                                   unsigned short* __restrict__ Abf) {
  const int row = blockIdx.x;                       // 0..B*S-1
  const int b   = row >> 11;
  const float* src  = Sc + (long)row * 2048;
  const int*   mrow = mask + (long)b * 2048;
  const int t = threadIdx.x;

  float v[8];
  float mx = -3.402823466e38f;
  #pragma unroll
  for (int i = 0; i < 8; ++i) {
    const int n = t + i * 256;
    float x = src[n];
    x = (mrow[n] == 0) ? -__builtin_inff() : x;
    v[i] = x;
    mx = fmaxf(mx, x);
  }
  __shared__ float red[256];
  red[t] = mx; __syncthreads();
  for (int s = 128; s > 0; s >>= 1) { if (t < s) red[t] = fmaxf(red[t], red[t + s]); __syncthreads(); }
  mx = red[0]; __syncthreads();

  float sum = 0.f;
  #pragma unroll
  for (int i = 0; i < 8; ++i) { v[i] = __expf(v[i] - mx); sum += v[i]; }
  red[t] = sum; __syncthreads();
  for (int s = 128; s > 0; s >>= 1) { if (t < s) red[t] += red[t + s]; __syncthreads(); }
  const float inv = 1.f / red[0];

  unsigned short* dst = Abf + (long)row * 2048;
  #pragma unroll
  for (int i = 0; i < 8; ++i) dst[t + i * 256] = f2bfu(v[i] * inv);
}

// ---------------------------------------------------------------------------
// Prep kernels: f32 -> bf16 convert, and f32 transpose -> bf16 (N-major wt).
// ---------------------------------------------------------------------------
__global__ __launch_bounds__(256) void gau_cvt_bf16(const float* __restrict__ src,
                                                    unsigned short* __restrict__ dst) {
  const long i = ((long)blockIdx.x * 256 + threadIdx.x);
  const float4 f0 = *(const float4*)(src + i * 8);
  const float4 f1 = *(const float4*)(src + i * 8 + 4);
  union { unsigned short s[8]; u32x4 q; } o;
  o.s[0] = f2bfu(f0.x); o.s[1] = f2bfu(f0.y); o.s[2] = f2bfu(f0.z); o.s[3] = f2bfu(f0.w);
  o.s[4] = f2bfu(f1.x); o.s[5] = f2bfu(f1.y); o.s[6] = f2bfu(f1.z); o.s[7] = f2bfu(f1.w);
  *(u32x4*)(dst + i * 8) = o.q;
}

__global__ __launch_bounds__(256) void gau_transpose_bf16(const float* __restrict__ src,
                                                          unsigned short* __restrict__ dst,
                                                          int rows, int cols) {
  __shared__ float t[32][33];
  const int c0 = blockIdx.x * 32, r0 = blockIdx.y * 32;
  const int tx = threadIdx.x, ty = threadIdx.y;     // block (32,8)
  #pragma unroll
  for (int i = 0; i < 4; ++i)
    t[ty + 8 * i][tx] = src[(long)(r0 + ty + 8 * i) * cols + c0 + tx];
  __syncthreads();
  #pragma unroll
  for (int i = 0; i < 4; ++i)
    dst[(long)(c0 + ty + 8 * i) * rows + r0 + tx] = f2bfu(t[tx][ty + 8 * i]);
}

// ---------------------------------------------------------------------------
extern "C" void kernel_launch(void* const* d_in, const int* in_sizes, int n_in,
                              void* d_out, int out_size, void* d_ws, size_t ws_size,
                              hipStream_t stream) {
  const float* x    = (const float*)d_in[0];
  const int*   mask = (const int*)  d_in[1];
  const float* W_i  = (const float*)d_in[2];
  const float* b_i  = (const float*)d_in[3];
  const float* q_w  = (const float*)d_in[4];
  const float* q_b  = (const float*)d_in[5];
  const float* k_w  = (const float*)d_in[6];
  const float* k_b  = (const float*)d_in[7];
  const float* W_o  = (const float*)d_in[8];
  const float* b_o  = (const float*)d_in[9];
  float* out = (float*)d_out;

  char* ws = (char*)d_ws; size_t off = 0;
  auto carve = [&](size_t bytes) -> void* {
    void* p = ws + off; off += (bytes + 255) & ~(size_t)255; return p;
  };
  unsigned short* Xbf = (unsigned short*)carve(16384L * 1024 * 2);
  unsigned short* WiT = (unsigned short*)carve(4224L  * 1024 * 2);
  unsigned short* WoT = (unsigned short*)carve(1024L  * 2048 * 2);
  unsigned short* U   = (unsigned short*)carve(16384L * 2048 * 2);
  unsigned short* Vt  = (unsigned short*)carve(8L * 2048 * 2048 * 2);
  unsigned short* Qb  = (unsigned short*)carve(16384L * 128 * 2);
  unsigned short* Kb  = (unsigned short*)carve(16384L * 128 * 2);
  float*          Sc  = (float*)         carve(8L * 2048 * 2048 * 4);
  unsigned short* Abf = (unsigned short*)carve(8L * 2048 * 2048 * 2);
  unsigned short* Obf = (unsigned short*)carve(16384L * 2048 * 2);

  // prep: x -> bf16, W_i^T, W_o^T (bf16, N-major for the GEMM B operand)
  gau_cvt_bf16<<<dim3(16384 * 1024 / (256 * 8)), 256, 0, stream>>>(x, Xbf);
  gau_transpose_bf16<<<dim3(4224 / 32, 1024 / 32), dim3(32, 8), 0, stream>>>(W_i, WiT, 1024, 4224);
  gau_transpose_bf16<<<dim3(1024 / 32, 2048 / 32), dim3(32, 8), 0, stream>>>(W_o, WoT, 2048, 1024);

  // K1: h = x @ W_i + b_i, SiLU, split into U / V^T / Q / K
  GemmP p1{}; p1.A = Xbf; p1.B = WiT; p1.K = 1024; p1.lda = 1024; p1.ldb = 1024;
  p1.bias = b_i; p1.qw = q_w; p1.qb = q_b; p1.kw = k_w; p1.kb = k_b;
  p1.u_out = U; p1.vt_out = Vt; p1.q_out = Qb; p1.k_out = Kb;
  gau_gemm<0><<<dim3(33, 128, 1), 256, 0, stream>>>(p1);

  // K2: scores = Q @ K^T / sqrt(128)  (per batch)
  GemmP p2{}; p2.A = Qb; p2.B = Kb; p2.sA = 2048L * 128; p2.sB = 2048L * 128;
  p2.K = 128; p2.lda = 128; p2.ldb = 128;
  p2.outf = Sc; p2.sOutF = 2048L * 2048; p2.ldf = 2048;
  p2.scale = 0.08838834764831843f;                  // 1/sqrt(128)
  gau_gemm<1><<<dim3(16, 16, 8), 256, 0, stream>>>(p2);

  // K3: masked row softmax -> bf16 A
  gau_softmax<<<dim3(16384), 256, 0, stream>>>(Sc, mask, Abf);

  // K4: o = u * (A @ v)  (per batch, B operand = V^T)
  GemmP p4{}; p4.A = Abf; p4.B = Vt; p4.sA = 2048L * 2048; p4.sB = 2048L * 2048;
  p4.K = 2048; p4.lda = 2048; p4.ldb = 2048;
  p4.outbf = Obf; p4.sOutBF = 2048L * 2048;
  p4.gate = U; p4.sGate = 2048L * 2048;
  gau_gemm<2><<<dim3(16, 16, 8), 256, 0, stream>>>(p4);

  // K5: out = o @ W_o + b_o
  GemmP p5{}; p5.A = Obf; p5.B = WoT; p5.K = 2048; p5.lda = 2048; p5.ldb = 2048;
  p5.outf = out; p5.ldf = 1024; p5.bias = b_o;
  gau_gemm<3><<<dim3(8, 128, 1), 256, 0, stream>>>(p5);
}
